// Attention_884763263593
// MI455X (gfx1250) — matrix operand order repaired and checked by reading, not compiled, not run on hardware
//
#include <hip/hip_runtime.h>
#include <hip/hip_bf16.h>
#include <cstdint>

// Problem constants (match reference)
#define B_    2
#define N_    2048
#define C_    768
#define H_    12
#define HD_   64
#define BH_   (B_ * H_)      // 24
#define M_TOT (B_ * N_)      // 4096
#define KQKV  (3 * C_)       // 2304

typedef __bf16 bf16_t;
typedef __attribute__((ext_vector_type(4)))  __bf16 v4bf;
typedef __attribute__((ext_vector_type(8)))  __bf16 v8bf;
typedef __attribute__((ext_vector_type(16))) __bf16 v16bf;
typedef __attribute__((ext_vector_type(8)))  float  v8f;
typedef __attribute__((ext_vector_type(4)))  unsigned int v4u;
typedef __attribute__((ext_vector_type(4)))  int v4i;
typedef __attribute__((ext_vector_type(8)))  int v8i;

static __device__ __forceinline__ v16bf cat16(v8bf lo, v8bf hi) {
  return __builtin_shufflevector(lo, hi, 0,1,2,3,4,5,6,7,8,9,10,11,12,13,14,15);
}
static __device__ __forceinline__ v8bf ld8(const bf16_t* p) {
  return *(const v8bf*)p;   // 16B aligned by construction
}
static __device__ __forceinline__ v8f wmma_bf16(v16bf a, v16bf b, v8f c) {
  return __builtin_amdgcn_wmma_f32_16x16x32_bf16(
      false, a, false, b, (short)0, c, false, false);
}
// LDS byte offset of a __shared__ object (flat addr low 32 bits = LDS offset)
static __device__ __forceinline__ uint32_t lds_off(const void* p) {
  return (uint32_t)(uintptr_t)p;
}

// ---------------------------------------------------------------------------
// Tensor Data Mover: 2D tile load Global->LDS (bf16 elements, data_size=2B).
// Pads pad_amount DWORDs into LDS after every pad_interval DWORDs so LDS row
// strides are bank-conflict free. Descriptor per CDNA5 ISA sec 8.3/8.4.
//   pad codes: interval 3 -> every 16 DW (64B), 4 -> every 32 DW (128B);
//              amount   3 -> 4 DW (16B).
// ---------------------------------------------------------------------------
static __device__ __forceinline__ void tdm_load_2d(
    uint64_t gaddr, uint32_t ldsoff,
    uint32_t tensor_d0, uint32_t tensor_d1, uint32_t stride0,
    uint32_t tile_d0, uint32_t tile_d1,
    uint32_t pad_interval_code, uint32_t pad_amount_code)
{
  v4u g0;
  g0[0] = 1u;                                               // count=1, user D#
  g0[1] = ldsoff;                                           // lds_addr (bytes)
  g0[2] = (uint32_t)gaddr;                                  // global_addr lo
  g0[3] = ((uint32_t)(gaddr >> 32) & 0x01FFFFFFu) | (2u << 30); // hi | type=2
  v8i g1;
  g1[0] = (int)((1u << 16) | (1u << 20) |                   // data_size=2B, pad_en
                (pad_interval_code << 22) | (pad_amount_code << 25));
  g1[1] = (int)((tensor_d0 & 0xFFFFu) << 16);               // dim0 -> bits 79:48
  g1[2] = (int)((tensor_d0 >> 16) | ((tensor_d1 & 0xFFFFu) << 16));
  g1[3] = (int)((tensor_d1 >> 16) | (tile_d0 << 16));       // tile_dim0 127:112
  g1[4] = (int)(tile_d1 & 0xFFFFu);                         // tile_dim1 (dim2=0)
  g1[5] = (int)stride0;                                     // dim0 stride lo32
  g1[6] = 0;                                                // stride hi, dim1 stride=0
  g1[7] = 0;
  v4i z4 = {};
#if defined(__clang_major__) && (__clang_major__ >= 23)
  v8i z8 = {};
  __builtin_amdgcn_tensor_load_to_lds(g0, g1, z4, z4, z8, 0);
#else
  __builtin_amdgcn_tensor_load_to_lds(g0, g1, z4, z4, 0);
#endif
}

// ---------------------------------------------------------------------------
// Kernel 0a: x (f32) -> Xbf (bf16), 4 elements / thread.
// ---------------------------------------------------------------------------
__global__ __launch_bounds__(256) void cvt_x_kernel(
    const float4* __restrict__ in, bf16_t* __restrict__ outp)
{
  const int i = blockIdx.x * 256 + threadIdx.x;
  const float4 v = in[i];
  v4bf o;
  o[0] = (bf16_t)v.x; o[1] = (bf16_t)v.y; o[2] = (bf16_t)v.z; o[3] = (bf16_t)v.w;
  *(v4bf*)(outp + 4 * (size_t)i) = o;
}

// ---------------------------------------------------------------------------
// Kernel 0b: weight transpose + convert: in f32 [R][Cc] -> out bf16 [Cc][R].
// ---------------------------------------------------------------------------
__global__ __launch_bounds__(256) void transpose_w_kernel(
    const float* __restrict__ in, bf16_t* __restrict__ outp, int R, int Cc)
{
  __shared__ float tile[32][33];
  const int bx = blockIdx.x * 32;           // column base
  const int by = blockIdx.y * 32;           // row base
  const int tx = threadIdx.x & 31, ty = threadIdx.x >> 5;
#pragma unroll
  for (int j = 0; j < 32; j += 8)
    tile[ty + j][tx] = in[(size_t)(by + ty + j) * Cc + bx + tx];
  __syncthreads();
#pragma unroll
  for (int j = 0; j < 32; j += 8)
    outp[(size_t)(bx + ty + j) * R + by + tx] = (bf16_t)tile[tx][ty + j];
}

// ---------------------------------------------------------------------------
// Kernel 1: qkv = Xbf @ Wqkv_t^T ; scatter into Q (scaled), K, V^T layouts.
// WG tile 256(M) x 64(N): 8 waves, each 32x64 => 8 WMMA / k-step, every B
// fragment reused twice. A (256x32) and B (64x32, [n][k]) TDM double-buffered;
// TDM padding gives 80B LDS rows (bank-conflict-free fragment reads).
// ---------------------------------------------------------------------------
__global__ __launch_bounds__(256) void qkv_gemm_kernel(
    const bf16_t* __restrict__ Xbf, const bf16_t* __restrict__ Wt,
    bf16_t* __restrict__ Qh, bf16_t* __restrict__ Kh, bf16_t* __restrict__ Vt)
{
  __shared__ __align__(16) bf16_t Alds[2][256][40];
  __shared__ __align__(16) bf16_t Blds[2][64][40];
  const int lane = threadIdx.x & 31;
  const int wv   = threadIdx.x >> 5;
  const int half = lane >> 4;
  const int l16  = lane & 15;
  const int Nbase = blockIdx.x * 64;
  const int Mbase = blockIdx.y * 256;
  const uint64_t xg = (uint64_t)(uintptr_t)Xbf;
  const uint64_t wg = (uint64_t)(uintptr_t)Wt;

  if (wv == 0) {
    tdm_load_2d(xg + 2ull * ((uint64_t)Mbase * C_), lds_off(&Alds[0][0][0]),
                C_, M_TOT, C_, 32, 256, 3, 3);
    tdm_load_2d(wg + 2ull * ((uint64_t)Nbase * C_), lds_off(&Blds[0][0][0]),
                C_, KQKV, C_, 32, 64, 3, 3);
  }

  v8f acc[2][4] = {};
  int buf = 0;
  for (int k0 = 0; k0 < C_; k0 += 32) {
    if (wv == 0) __builtin_amdgcn_s_wait_tensorcnt(0);
    __syncthreads();                      // tile[buf] visible; tile[buf^1] free
    if (wv == 0 && k0 + 32 < C_) {
      tdm_load_2d(xg + 2ull * ((uint64_t)Mbase * C_ + k0 + 32),
                  lds_off(&Alds[buf ^ 1][0][0]), C_, M_TOT, C_, 32, 256, 3, 3);
      tdm_load_2d(wg + 2ull * ((uint64_t)Nbase * C_ + k0 + 32),
                  lds_off(&Blds[buf ^ 1][0][0]), C_, KQKV, C_, 32, 64, 3, 3);
    }
    // Batch all fragment loads, then issue WMMAs back-to-back.
    const bf16_t* ar0 = &Alds[buf][wv * 32 + l16][0];
    const bf16_t* ar1 = &Alds[buf][wv * 32 + 16 + l16][0];
    const v16bf a0 = cat16(ld8(ar0 + half * 8), ld8(ar0 + 16 + half * 8));
    const v16bf a1 = cat16(ld8(ar1 + half * 8), ld8(ar1 + 16 + half * 8));
    v16bf bfr[4];
#pragma unroll
    for (int t = 0; t < 4; ++t) {
      const bf16_t* bp = &Blds[buf][t * 16 + l16][half * 8];
      bfr[t] = cat16(ld8(bp), ld8(bp + 16));
    }
#pragma unroll
    for (int t = 0; t < 4; ++t) {
      acc[0][t] = wmma_bf16(a0, bfr[t], acc[0][t]);
      acc[1][t] = wmma_bf16(a1, bfr[t], acc[1][t]);
    }
    buf ^= 1;
  }

  // Epilogue: column c -> (which, head, d); row r -> (batch, n)
#pragma unroll
  for (int mi = 0; mi < 2; ++mi) {
#pragma unroll
    for (int t = 0; t < 4; ++t) {
      const int c = Nbase + t * 16 + l16;
      const int which = (c >= 2 * C_) ? 2 : (c >= C_ ? 1 : 0);
      const int cc = c - which * C_;
      const int h = cc >> 6, d = cc & 63;
#pragma unroll
      for (int j = 0; j < 8; ++j) {
        const int r = Mbase + wv * 32 + mi * 16 + j + 8 * half; // C-layout row
        const int b = r >> 11, n = r & (N_ - 1);
        const int bh = b * H_ + h;
        const float v = acc[mi][t][j];
        if (which == 0)      Qh[((size_t)bh * N_ + n) * HD_ + d] = (bf16_t)(v * 0.125f);
        else if (which == 1) Kh[((size_t)bh * N_ + n) * HD_ + d] = (bf16_t)v;
        else                 Vt[((size_t)bh * HD_ + d) * N_ + n] = (bf16_t)v;
      }
    }
  }
}

// ---------------------------------------------------------------------------
// Kernel 2: flash attention. 8 waves/WG share one (b,h); each wave owns a
// 16-row Q block. K (32x64) and V^T (64x32) tiles TDM double-buffered into
// LDS once per WG per 32 keys. Online softmax via shfl_xor row reductions;
// P re-laid out through per-wave LDS (s_wait_dscnt fence).
// ---------------------------------------------------------------------------
__global__ __launch_bounds__(256) void attn_kernel(
    const bf16_t* __restrict__ Qh, const bf16_t* __restrict__ Kh,
    const bf16_t* __restrict__ Vt, bf16_t* __restrict__ Ob)
{
  __shared__ __align__(16) bf16_t Klds[2][32][72];   // 128B rows + 16B pad
  __shared__ __align__(16) bf16_t Vlds[2][64][40];   // 64B rows + 16B pad
  __shared__ __align__(16) bf16_t Plds[8][16 * 40];  // per-wave P (16x32)
  const int lane = threadIdx.x & 31;
  const int wv   = threadIdx.x >> 5;
  const int half = lane >> 4;
  const int l16  = lane & 15;
  const int bh   = blockIdx.x >> 4;                  // 24 (b,h) pairs
  const int qblk = (blockIdx.x & 15) * 8 + wv;       // 128 q-blocks per bh
  const int b = bh / H_, h = bh % H_;

  const uint64_t kg = (uint64_t)(uintptr_t)(Kh + (size_t)bh * N_ * HD_);
  const uint64_t vg = (uint64_t)(uintptr_t)(Vt + (size_t)bh * HD_ * N_);

  // Q fragments (16x64 = two 16x32 A fragments), Q pre-scaled by HD^-0.5
  const bf16_t* qrow = Qh + ((size_t)bh * N_ + qblk * 16 + l16) * HD_;
  const v16bf aq0 = cat16(ld8(qrow + half * 8),      ld8(qrow + 16 + half * 8));
  const v16bf aq1 = cat16(ld8(qrow + 32 + half * 8), ld8(qrow + 48 + half * 8));

  if (wv == 0) {
    tdm_load_2d(kg, lds_off(&Klds[0][0][0]), HD_, N_, HD_, HD_, 32, 4, 3);
    tdm_load_2d(vg, lds_off(&Vlds[0][0][0]), N_, HD_, N_, 32, HD_, 3, 3);
  }

  v8f o[4] = {};
  float Mrow[8], Lrow[8];
#pragma unroll
  for (int j = 0; j < 8; ++j) { Mrow[j] = -3.0e38f; Lrow[j] = 0.0f; }

  bf16_t* pl = &Plds[wv][0];
  int buf = 0;
  for (int kt = 0; kt < N_ / 32; ++kt) {
    if (wv == 0) __builtin_amdgcn_s_wait_tensorcnt(0);
    __syncthreads();
    if (wv == 0 && kt + 1 < N_ / 32) {
      tdm_load_2d(kg + 2ull * ((uint64_t)(kt + 1) * 32 * HD_),
                  lds_off(&Klds[buf ^ 1][0][0]), HD_, N_, HD_, HD_, 32, 4, 3);
      tdm_load_2d(vg + 2ull * ((uint64_t)(kt + 1) * 32),
                  lds_off(&Vlds[buf ^ 1][0][0]), N_, HD_, N_, 32, HD_, 3, 3);
    }

    // ---- S = Q K^T for 32 keys: batch 4 K fragments, then 4 WMMAs ----
    v8f s0, s1;
    {
      const bf16_t* kr0 = &Klds[buf][l16][half * 8];
      const bf16_t* kr1 = &Klds[buf][16 + l16][half * 8];
      const v16bf k00 = cat16(ld8(kr0),      ld8(kr0 + 16));
      const v16bf k01 = cat16(ld8(kr0 + 32), ld8(kr0 + 32 + 16));
      const v16bf k10 = cat16(ld8(kr1),      ld8(kr1 + 16));
      const v16bf k11 = cat16(ld8(kr1 + 32), ld8(kr1 + 32 + 16));
      const v8f z = {};
      s0 = wmma_bf16(aq0, k00, z);
      s1 = wmma_bf16(aq0, k10, z);
      s0 = wmma_bf16(aq1, k01, s0);
      s1 = wmma_bf16(aq1, k11, s1);
    }

    // ---- online softmax; each row lives across 16 lanes of a half-wave ----
#pragma unroll
    for (int j = 0; j < 8; ++j) {
      float mx = fmaxf(s0[j], s1[j]);
#pragma unroll
      for (int off = 1; off < 16; off <<= 1) mx = fmaxf(mx, __shfl_xor(mx, off, 32));
      const float mnew  = fmaxf(Mrow[j], mx);
      const float alpha = __expf(Mrow[j] - mnew);
      const float p0 = __expf(s0[j] - mnew);
      const float p1 = __expf(s1[j] - mnew);
      float rs = p0 + p1;
#pragma unroll
      for (int off = 1; off < 16; off <<= 1) rs += __shfl_xor(rs, off, 32);
      Lrow[j] = Lrow[j] * alpha + rs;
      Mrow[j] = mnew;
      o[0][j] *= alpha; o[1][j] *= alpha; o[2][j] *= alpha; o[3][j] *= alpha;
      const int m = j + 8 * half;                 // C-layout row -> P row
      pl[m * 40 + l16]      = (bf16_t)p0;
      pl[m * 40 + 16 + l16] = (bf16_t)p1;
    }
    asm volatile("s_wait_dscnt 0" ::: "memory");  // LDS RAW fence (same wave)

    // ---- P fragment (A layout) from LDS ----
    const bf16_t* pr = pl + l16 * 40;
    const v16bf pf = cat16(ld8(pr + half * 8), ld8(pr + 16 + half * 8));

    // ---- O += P(16x32) @ V(32x64): batch 4 V fragments, then 4 WMMAs ----
    {
      v16bf vf[4];
#pragma unroll
      for (int t = 0; t < 4; ++t) {
        const bf16_t* vr = &Vlds[buf][t * 16 + l16][half * 8];
        vf[t] = cat16(ld8(vr), ld8(vr + 16));
      }
#pragma unroll
      for (int t = 0; t < 4; ++t) o[t] = wmma_bf16(pf, vf[t], o[t]);
    }
    buf ^= 1;
  }

  // ---- normalize and store O in [B*N, C] bf16 for the proj GEMM ----
#pragma unroll
  for (int j = 0; j < 8; ++j) {
    const float inv = 1.0f / Lrow[j];
    const int m = j + 8 * half;
    const int r = b * N_ + qblk * 16 + m;
    const size_t base = (size_t)r * C_ + h * HD_;
#pragma unroll
    for (int t = 0; t < 4; ++t)
      Ob[base + t * 16 + l16] = (bf16_t)(o[t][j] * inv);
  }
}

// ---------------------------------------------------------------------------
// Kernel 3: out = Ob(bf16) @ Wproj_t^T + b_proj, f32 output. Same tiling as K1.
// ---------------------------------------------------------------------------
__global__ __launch_bounds__(256) void proj_gemm_kernel(
    const bf16_t* __restrict__ A, const bf16_t* __restrict__ Wt,
    const float* __restrict__ bias, float* __restrict__ outp)
{
  __shared__ __align__(16) bf16_t Alds[2][256][40];
  __shared__ __align__(16) bf16_t Blds[2][64][40];
  const int lane = threadIdx.x & 31;
  const int wv   = threadIdx.x >> 5;
  const int half = lane >> 4;
  const int l16  = lane & 15;
  const int Nbase = blockIdx.x * 64;
  const int Mbase = blockIdx.y * 256;
  const uint64_t ag = (uint64_t)(uintptr_t)A;
  const uint64_t wg = (uint64_t)(uintptr_t)Wt;

  if (wv == 0) {
    tdm_load_2d(ag + 2ull * ((uint64_t)Mbase * C_), lds_off(&Alds[0][0][0]),
                C_, M_TOT, C_, 32, 256, 3, 3);
    tdm_load_2d(wg + 2ull * ((uint64_t)Nbase * C_), lds_off(&Blds[0][0][0]),
                C_, C_, C_, 32, 64, 3, 3);
  }

  v8f acc[2][4] = {};
  int buf = 0;
  for (int k0 = 0; k0 < C_; k0 += 32) {
    if (wv == 0) __builtin_amdgcn_s_wait_tensorcnt(0);
    __syncthreads();
    if (wv == 0 && k0 + 32 < C_) {
      tdm_load_2d(ag + 2ull * ((uint64_t)Mbase * C_ + k0 + 32),
                  lds_off(&Alds[buf ^ 1][0][0]), C_, M_TOT, C_, 32, 256, 3, 3);
      tdm_load_2d(wg + 2ull * ((uint64_t)Nbase * C_ + k0 + 32),
                  lds_off(&Blds[buf ^ 1][0][0]), C_, C_, C_, 32, 64, 3, 3);
    }
    const bf16_t* ar0 = &Alds[buf][wv * 32 + l16][0];
    const bf16_t* ar1 = &Alds[buf][wv * 32 + 16 + l16][0];
    const v16bf a0 = cat16(ld8(ar0 + half * 8), ld8(ar0 + 16 + half * 8));
    const v16bf a1 = cat16(ld8(ar1 + half * 8), ld8(ar1 + 16 + half * 8));
    v16bf bfr[4];
#pragma unroll
    for (int t = 0; t < 4; ++t) {
      const bf16_t* bp = &Blds[buf][t * 16 + l16][half * 8];
      bfr[t] = cat16(ld8(bp), ld8(bp + 16));
    }
#pragma unroll
    for (int t = 0; t < 4; ++t) {
      acc[0][t] = wmma_bf16(a0, bfr[t], acc[0][t]);
      acc[1][t] = wmma_bf16(a1, bfr[t], acc[1][t]);
    }
    buf ^= 1;
  }

#pragma unroll
  for (int mi = 0; mi < 2; ++mi) {
#pragma unroll
    for (int t = 0; t < 4; ++t) {
      const int c = Nbase + t * 16 + l16;
      const float bi = bias[c];
#pragma unroll
      for (int j = 0; j < 8; ++j) {
        const int r = Mbase + wv * 32 + mi * 16 + j + 8 * half;
        outp[(size_t)r * C_ + c] = acc[mi][t][j] + bi;
      }
    }
  }
}

// ---------------------------------------------------------------------------
extern "C" void kernel_launch(void* const* d_in, const int* in_sizes, int n_in,
                              void* d_out, int out_size, void* d_ws, size_t ws_size,
                              hipStream_t stream)
{
  (void)in_sizes; (void)n_in; (void)out_size; (void)ws_size;
  const float* x      = (const float*)d_in[0];
  const float* w_qkv  = (const float*)d_in[1];
  const float* w_proj = (const float*)d_in[2];
  const float* b_proj = (const float*)d_in[3];
  float* outp = (float*)d_out;

  char* ws = (char*)d_ws;
  size_t cur = 0;
  bf16_t* Xbf = (bf16_t*)(ws + cur); cur += (size_t)M_TOT * C_ * 2;   // [4096][768]
  bf16_t* Wqt = (bf16_t*)(ws + cur); cur += (size_t)KQKV * C_ * 2;    // [2304][768]
  bf16_t* Wpt = (bf16_t*)(ws + cur); cur += (size_t)C_ * C_ * 2;      // [768][768]
  bf16_t* Qh  = (bf16_t*)(ws + cur); cur += (size_t)BH_ * N_ * HD_ * 2;
  bf16_t* Kh  = (bf16_t*)(ws + cur); cur += (size_t)BH_ * N_ * HD_ * 2;
  bf16_t* Vt  = (bf16_t*)(ws + cur); cur += (size_t)BH_ * N_ * HD_ * 2;
  bf16_t* Ob  = (bf16_t*)(ws + cur); cur += (size_t)M_TOT * C_ * 2;

  cvt_x_kernel<<<dim3(M_TOT * C_ / 1024), 256, 0, stream>>>((const float4*)x, Xbf);
  transpose_w_kernel<<<dim3(KQKV / 32, C_ / 32), 256, 0, stream>>>(w_qkv, Wqt, C_, KQKV);
  transpose_w_kernel<<<dim3(C_ / 32,  C_ / 32), 256, 0, stream>>>(w_proj, Wpt, C_, C_);
  qkv_gemm_kernel<<<dim3(KQKV / 64, M_TOT / 256), 256, 0, stream>>>(Xbf, Wqt, Qh, Kh, Vt);
  attn_kernel<<<dim3(BH_ * 16), 256, 0, stream>>>(Qh, Kh, Vt, Ob);
  proj_gemm_kernel<<<dim3(C_ / 64, M_TOT / 256), 256, 0, stream>>>(Ob, Wpt, b_proj, outp);
}
